// SkipGram_27857157882110
// MI455X (gfx1250) — matrix operand order repaired and checked by reading, not compiled, not run on hardware
//
#include <hip/hip_runtime.h>

#define VOCAB 50000
#define EMB   128
#define BATCH 2048
#define NNEG  20

typedef __attribute__((ext_vector_type(4)))  float    v4f;
typedef __attribute__((ext_vector_type(16))) _Float16 v16h;
typedef __attribute__((ext_vector_type(8)))  float    v8f;

// ---------------------------------------------------------------------------
// Kernel 1: recover the one-hot index (and value) from each row of cen_word.
// 400 MB streamed once -> dominant cost (~17us at 23.3 TB/s). Non-temporal
// b128 loads: no reuse, don't pollute L2.
// grid = (BATCH, 4): 4 segments of 3125 float4 per 50000-float row.
// ---------------------------------------------------------------------------
__global__ void __launch_bounds__(256) onehot_extract(const float* __restrict__ cen,
                                                      int*   __restrict__ outIdx,
                                                      float* __restrict__ outVal) {
  const int b     = blockIdx.x;
  const int start = blockIdx.y * 3125;            // in float4 units
  const v4f* row  = (const v4f*)(cen + (size_t)b * VOCAB);

  int fI = -1; float fV = 0.0f;
  for (int i = start + (int)threadIdx.x; i < start + 3125; i += 256) {
    v4f v = __builtin_nontemporal_load(row + i);
    if      (v.x != 0.0f) { fI = 4*i + 0; fV = v.x; }
    else if (v.y != 0.0f) { fI = 4*i + 1; fV = v.y; }
    else if (v.z != 0.0f) { fI = 4*i + 2; fV = v.z; }
    else if (v.w != 0.0f) { fI = 4*i + 3; fV = v.w; }
  }
  // Exactly one nonzero per one-hot row -> at most one thread writes.
  if (fI >= 0) { outIdx[b] = fI; outVal[b] = fV; }
}

__device__ __forceinline__ float log_sigmoid(float x) {
  // numerically stable: min(x,0) - log1p(exp(-|x|))
  return fminf(x, 0.0f) - log1pf(expf(-fabsf(x)));
}

// ---------------------------------------------------------------------------
// Kernel 2: one wave per batch element. 21x128 gathered GEMV via
// v_wmma_f32_16x16x32_f16 (2 row tiles x 4 K-chunks), then log-sigmoid and
// global atomic reduction of the negated loss.
// grid = BATCH/8 blocks x 256 threads (8 waves).
// ---------------------------------------------------------------------------
__global__ void __launch_bounds__(256) sgns_loss(const int*   __restrict__ con,
                                                 const int*   __restrict__ neg,
                                                 const float* __restrict__ in_embed,   // [D, V]
                                                 const float* __restrict__ out_embed,  // [V, D]
                                                 const int*   __restrict__ cenIdx,
                                                 const float* __restrict__ cenVal,
                                                 float*       __restrict__ loss) {
  __shared__ float sH[8][EMB];

  const int lane = threadIdx.x & 31;
  const int w    = threadIdx.x >> 5;
  const int b    = blockIdx.x * 8 + w;            // 256*8 == 2048, no tail

  // --- h[b][d] = val * in_embed[d, idx]  (column gather, stride VOCAB) -----
  const int   ci = cenIdx[b];
  const float cv = cenVal[b];
  #pragma unroll
  for (int k = 0; k < 4; ++k) {
    const int d = lane * 4 + k;
    sH[w][d] = cv * in_embed[(size_t)d * VOCAB + ci];
  }
  __syncthreads();

  // --- word index per lane: lane 0 = positive, lanes 1..20 = negatives ----
  int myIdx = 0;
  if (lane == 0)          myIdx = con[b];
  else if (lane <= NNEG)  myIdx = neg[b * NNEG + (lane - 1)];

  // --- B fragments: 32x16 f16, only column N=0 is nonzero (holds h chunk).
  // Layout: lanes 0-15 -> N=lane, K=base+0..7,16..23; lanes 16-31 -> N=lane-16,
  // K=base+8..15,24..31 (same K order as the A fragments).
  v16h bfrag[4];
  #pragma unroll
  for (int kk = 0; kk < 4; ++kk) {
    v16h bt = {};
    if (lane == 0) {
      #pragma unroll
      for (int i = 0; i < 16; ++i) bt[i] = (_Float16)sH[w][kk * 32 + (i < 8 ? i : i + 8)];
    } else if (lane == 16) {
      #pragma unroll
      for (int i = 0; i < 16; ++i) bt[i] = (_Float16)sH[w][kk * 32 + (i < 8 ? 8 + i : 16 + i)];
    }
    bfrag[kk] = bt;
  }

  // --- A fragments + WMMA: A row M = lane&15 of tile t (global j = t*16+M).
  // 16-bit A 16x32 layout: lane group 0 holds K = base+0..7 and base+16..23,
  // group 1 holds K = base+8..15 and base+24..31.
  const int g = lane >> 4;
  const int m = lane & 15;

  v8f acc[2] = { {}, {} };
  #pragma unroll
  for (int t = 0; t < 2; ++t) {
    const int j  = t * 16 + m;                       // gathered-row id 0..31
    const int wj = __shfl(myIdx, j, 32);             // out_embed row (valid if j<=20)
    const float* rp = out_embed + (size_t)wj * EMB;
    #pragma unroll
    for (int kk = 0; kk < 4; ++kk) {
      v16h a = {};
      if (j <= NNEG) {
        const v4f* p   = (const v4f*)rp;
        const int  s0  = (kk * 32 + g * 8) >> 2;     // float4 index of 1st segment
        const int  s1  = s0 + 4;                     // +16 floats -> 2nd segment
        v4f x0 = p[s0], x1 = p[s0 + 1];
        v4f x2 = p[s1], x3 = p[s1 + 1];
        a[0]=(_Float16)x0.x; a[1]=(_Float16)x0.y; a[2] =(_Float16)x0.z; a[3] =(_Float16)x0.w;
        a[4]=(_Float16)x1.x; a[5]=(_Float16)x1.y; a[6] =(_Float16)x1.z; a[7] =(_Float16)x1.w;
        a[8]=(_Float16)x2.x; a[9]=(_Float16)x2.y; a[10]=(_Float16)x2.z; a[11]=(_Float16)x2.w;
        a[12]=(_Float16)x3.x;a[13]=(_Float16)x3.y; a[14]=(_Float16)x3.z; a[15]=(_Float16)x3.w;
      }
      // EXEC is all-ones here (branch above is closed, waves fully active).
      acc[t] = __builtin_amdgcn_wmma_f32_16x16x32_f16(
          /*neg_a=*/false, a, /*neg_b=*/false, bfrag[kk],
          /*c_mod=*/(short)0, acc[t], /*reuse_a=*/false, /*reuse_b=*/false);
    }
  }

  // --- scores live in column N=0: lane 0 VGPR r -> M=r (j=r / j=16+r),
  //     lane 16 VGPR r -> M=8+r (j=8+r). Valid j: 0..20.
  float part = 0.0f;
  if (lane == 0) {
    part += log_sigmoid(acc[0][0]);                        // j=0 positive
    #pragma unroll
    for (int r = 1; r < 8; ++r) part += log_sigmoid(-acc[0][r]);   // j=1..7
    #pragma unroll
    for (int r = 0; r < 5; ++r) part += log_sigmoid(-acc[1][r]);   // j=16..20
  } else if (lane == 16) {
    #pragma unroll
    for (int r = 0; r < 8; ++r) part += log_sigmoid(-acc[0][r]);   // j=8..15
  }
  part += __shfl_down(part, 16, 32);
  if (lane == 0) atomicAdd(loss, -part);
}

// ---------------------------------------------------------------------------
extern "C" void kernel_launch(void* const* d_in, const int* in_sizes, int n_in,
                              void* d_out, int out_size, void* d_ws, size_t ws_size,
                              hipStream_t stream) {
  const float* cen       = (const float*)d_in[0];   // [B, V] one-hot
  const int*   con       = (const int*)  d_in[1];   // [B]
  const int*   neg       = (const int*)  d_in[2];   // [B, K]
  const float* in_embed  = (const float*)d_in[3];   // [D, V]
  const float* out_embed = (const float*)d_in[4];   // [V, D]
  float* loss = (float*)d_out;

  int*   wsIdx = (int*)d_ws;
  float* wsVal = (float*)((char*)d_ws + BATCH * sizeof(int));

  hipMemsetAsync(d_out, 0, sizeof(float), stream);  // graph-capture safe

  onehot_extract<<<dim3(BATCH, 4), 256, 0, stream>>>(cen, wsIdx, wsVal);
  sgns_loss<<<BATCH / 8, 256, 0, stream>>>(con, neg, in_embed, out_embed,
                                           wsIdx, wsVal, loss);
}
